// CrossViewSwapAttention_41652592836743
// MI455X (gfx1250) — matrix operand order repaired and checked by reading, not compiled, not run on hardware
//
#include <hip/hip_runtime.h>
#include <math.h>
#include <stdint.h>

typedef _Float16 h16;
typedef __attribute__((ext_vector_type(16))) _Float16 v16h;
typedef __attribute__((ext_vector_type(8)))  float    v8f;
typedef __attribute__((ext_vector_type(4)))  uint32_t u32x4;
typedef __attribute__((ext_vector_type(8)))  uint32_t u32x8;

#define DEV static __device__ __forceinline__

// ---------------------------------------------------------------------------
// WMMA helper: D = A(16x32 f16) x B(32x16 f16) + C(16x16 f32)
// ---------------------------------------------------------------------------
DEV v8f wmma16(v16h a, v16h b, v8f c) {
  return __builtin_amdgcn_wmma_f32_16x16x32_f16(false, a, false, b, (short)0, c,
                                                false, false);
}

// A-matrix per-lane K index (contiguous order, matches B fills):
// lanes 0-15 hold K {0..15}; lanes 16-31 hold K {16..31}
DEV int kmapA(int lane, int j) { return j + ((lane & 16) ? 16 : 0); }

// Reductions confined to 16-lane halves (wave32)
DEV float redmax16(float v) {
#pragma unroll
  for (int m = 1; m < 16; m <<= 1) v = fmaxf(v, __shfl_xor(v, m, 32));
  return v;
}
DEV float redsum16(float v) {
#pragma unroll
  for (int m = 1; m < 16; m <<= 1) v += __shfl_xor(v, m, 32);
  return v;
}
DEV float redsum32(float v) {
#pragma unroll
  for (int m = 1; m < 32; m <<= 1) v += __shfl_xor(v, m, 32);
  return v;
}

// ---------------------------------------------------------------------------
// Tensor Data Mover descriptor issue (2D, groups 0+1 only).
// D# bitfields per CDNA5 ISA. data_size=2B.
// ---------------------------------------------------------------------------
DEV void tdm_load_2d(const h16* gsrc, uint32_t lds_off, uint32_t tensor_d0,
                     uint32_t tensor_d1, uint32_t tile_d0, uint32_t tile_d1,
                     uint64_t stride0) {
  uint64_t ga = (uint64_t)(uintptr_t)gsrc;
  u32x4 g0;
  g0[0] = 1u;                  // count=1, user mode
  g0[1] = lds_off;             // lds_addr (bytes)
  g0[2] = (uint32_t)ga;        // global_addr[31:0]
  g0[3] = ((uint32_t)(ga >> 32) & 0x01FFFFFFu) | (2u << 30);  // addr[56:32]|type=2
  u32x8 g1;
  g1[0] = 1u << 16;            // data_size = 1 (2 bytes)
  g1[1] = (tensor_d0 & 0xFFFFu) << 16;
  g1[2] = ((tensor_d0 >> 16) & 0xFFFFu) | ((tensor_d1 & 0xFFFFu) << 16);
  g1[3] = ((tensor_d1 >> 16) & 0xFFFFu) | ((tile_d0 & 0xFFFFu) << 16);
  g1[4] = (tile_d1 & 0xFFFFu); // tile_dim2 = 0
  g1[5] = (uint32_t)stride0;
  g1[6] = (uint32_t)((stride0 >> 32) & 0xFFFFu);  // dim1_stride = 0 (2D)
  g1[7] = 0u;
  asm volatile("tensor_load_to_lds %0, %1" :: "s"(g0), "s"(g1) : "memory");
}

// 3D variant (groups 0..3): x stride 1, y stride stride0, z stride stride1.
DEV void tdm_load_3d(const h16* gsrc, uint32_t lds_off, uint32_t d0, uint32_t d1,
                     uint32_t d2, uint32_t t0, uint32_t t1, uint32_t t2,
                     uint64_t stride0, uint64_t stride1) {
  uint64_t ga = (uint64_t)(uintptr_t)gsrc;
  u32x4 g0;
  g0[0] = 1u;
  g0[1] = lds_off;
  g0[2] = (uint32_t)ga;
  g0[3] = ((uint32_t)(ga >> 32) & 0x01FFFFFFu) | (2u << 30);
  u32x8 g1;
  g1[0] = 1u << 16;
  g1[1] = (d0 & 0xFFFFu) << 16;
  g1[2] = ((d0 >> 16) & 0xFFFFu) | ((d1 & 0xFFFFu) << 16);
  g1[3] = ((d1 >> 16) & 0xFFFFu) | ((t0 & 0xFFFFu) << 16);
  g1[4] = (t1 & 0xFFFFu) | ((t2 & 0xFFFFu) << 16);
  g1[5] = (uint32_t)stride0;
  g1[6] = (uint32_t)((stride0 >> 32) & 0xFFFFu) | ((uint32_t)(stride1 & 0xFFFFu) << 16);
  g1[7] = (uint32_t)(stride1 >> 16);
  u32x4 g2;
  g2[0] = d2;   // tensor_dim2
  g2[1] = 0u;   // tensor_dim3
  g2[2] = 0u;   // tensor_dim2_stride lo (4th dim unused)
  g2[3] = 0u;   // stride hi | tile_dim3=0
  u32x4 g3;
  g3[0] = 0u; g3[1] = 0u; g3[2] = 0u; g3[3] = 0u;
  asm volatile("tensor_load_to_lds %0, %1, %2, %3"
               :: "s"(g0), "s"(g1), "s"(g2), "s"(g3) : "memory");
}

// ---------------------------------------------------------------------------
// f32 -> f16 convert (weights)
// ---------------------------------------------------------------------------
__global__ void cvt_kernel(const float* __restrict__ s, h16* __restrict__ d, int n) {
  int i = blockIdx.x * 256 + threadIdx.x;
  if (i < n) d[i] = (h16)s[i];
}

// ---------------------------------------------------------------------------
// BN affine precompute: inv = g/sqrt(v+eps); beta = b - m*inv
// ---------------------------------------------------------------------------
__global__ void bnprep_kernel(const float* pg, const float* pb, const float* pm,
                              const float* pv, const float* lgm, const float* lb,
                              const float* lm, const float* lv2, float* invp,
                              float* betap, float* invl, float* betal) {
  int c = threadIdx.x;
  float ip = pg[c] * rsqrtf(pv[c] + 1e-5f);
  invp[c] = ip;
  betap[c] = pb[c] - pm[c] * ip;
  float il = lgm[c] * rsqrtf(lv2[c] + 1e-5f);
  invl[c] = il;
  betal[c] = lb[c] - lm[c] * il;
}

// ---------------------------------------------------------------------------
// Gather BEV query tokens in window order: qtok[(l,qpos)][c] = x[c][h][w]
// ---------------------------------------------------------------------------
__global__ void gatherq_kernel(const float* __restrict__ x, float* __restrict__ qtok) {
  int tok = blockIdx.x, c = threadIdx.x;
  int l = tok >> 8, qp2 = tok & 255;
  int h = (l >> 3) * 16 + (qp2 >> 4);
  int w = (l & 7) * 16 + (qp2 & 15);
  qtok[(size_t)tok * 256 + c] = x[(size_t)c * 16384 + h * 128 + w];
}

// ---------------------------------------------------------------------------
// LayerNorm over d=256, token-major f32 in -> f16 out
// ---------------------------------------------------------------------------
__global__ void ln_kernel(const float* __restrict__ in, const float* __restrict__ g,
                          const float* __restrict__ b, h16* __restrict__ out, int T) {
  int wave = threadIdx.x >> 5, lane = threadIdx.x & 31;
  int tok = blockIdx.x * 8 + wave;
  if (tok >= T) return;
  float v[8];
  float s = 0.f;
#pragma unroll
  for (int j = 0; j < 8; ++j) {
    v[j] = in[(size_t)tok * 256 + lane + 32 * j];
    s += v[j];
  }
  float mu = redsum32(s) * (1.f / 256.f);
  float q = 0.f;
#pragma unroll
  for (int j = 0; j < 8; ++j) {
    float d = v[j] - mu;
    q += d * d;
  }
  float rstd = rsqrtf(redsum32(q) * (1.f / 256.f) + 1e-5f);
#pragma unroll
  for (int j = 0; j < 8; ++j) {
    int c = lane + 32 * j;
    out[(size_t)tok * 256 + c] = (h16)((v[j] - mu) * rstd * g[c] + b[c]);
  }
}

// ---------------------------------------------------------------------------
// Fused conv1x1 projections over feature (token-major output). Feature tiles
// staged in LDS via async DMA (GLOBAL_LOAD_ASYNC_TO_LDS_B128, ASYNCcnt).
// ---------------------------------------------------------------------------
__global__ void feat_kernel(const float* __restrict__ feat, const h16* __restrict__ projH,
                            const h16* __restrict__ imgH, const h16* __restrict__ linH,
                            const float* __restrict__ invp, const float* __restrict__ betap,
                            const float* __restrict__ invl, const float* __restrict__ betal,
                            float* __restrict__ keyT, float* __restrict__ valT) {
  __shared__ float fs[32][16];
  int wave = threadIdx.x >> 5, lane = threadIdx.x & 31, tid = threadIdx.x;
  int lg = (lane & 16) ? 1 : 0, ln16 = lane & 15;
  int t0 = blockIdx.x * 16;
  int Mb = blockIdx.y * 128 + wave * 16;
  v8f ak = {}, av = {};
  for (int kb = 0; kb < 256; kb += 32) {
    __syncthreads();
    if (tid < 128) {  // 128 lanes x b128 async DMA -> 32x16 f32 tile
      int cin = tid >> 2, t4 = (tid & 3) * 4;
      int token = t0 + t4;
      int cam = token >> 12, rem = token & 4095;
      int l = rem >> 6, kp2 = rem & 63;
      int hh = (l >> 3) * 8 + (kp2 >> 3);
      int ww = (l & 7) * 8 + (kp2 & 7);
      uint64_t ga = (uint64_t)(uintptr_t)(feat +
          (size_t)(cam * 256 + kb + cin) * 4096 + hh * 64 + ww);
      uint32_t lo = (uint32_t)(uintptr_t)(&fs[cin][t4]);
      asm volatile("global_load_async_to_lds_b128 %0, %1, off"
                   :: "v"(lo), "v"(ga) : "memory");
    }
    asm volatile("s_wait_asynccnt 0x0" ::: "memory");
    __syncthreads();
    int arow = Mb + ln16;
    v16h ap, ai, al;
#pragma unroll
    for (int j = 0; j < 16; ++j) {
      int kk = kb + kmapA(lane, j);
      ap[j] = projH[(size_t)arow * 256 + kk];
      ai[j] = imgH[(size_t)arow * 256 + kk];
      al[j] = linH[(size_t)arow * 256 + kk];
    }
    v16h braw, bp, bl;
#pragma unroll
    for (int j = 0; j < 16; ++j) {
      int kk = lg * 16 + j;  // B-matrix K row for this lane group
      float raw = fs[kk][ln16];
      int cin = kb + kk;
      braw[j] = (h16)raw;
      bp[j] = (h16)fmaxf(raw * invp[cin] + betap[cin], 0.f);
      bl[j] = (h16)fmaxf(raw * invl[cin] + betal[cin], 0.f);
    }
    ak = wmma16(ap, bp, ak);
    ak = wmma16(ai, braw, ak);
    av = wmma16(al, bl, av);
  }
  int tokn = t0 + ln16;
  int chb = Mb + lg * 8;
  size_t base = (size_t)tokn * 256 + chb;
  ((float4*)(keyT + base))[0] = make_float4(ak[0], ak[1], ak[2], ak[3]);
  ((float4*)(keyT + base))[1] = make_float4(ak[4], ak[5], ak[6], ak[7]);
  ((float4*)(valT + base))[0] = make_float4(av[0], av[1], av[2], av[3]);
  ((float4*)(valT + base))[1] = make_float4(av[4], av[5], av[6], av[7]);
}

// ---------------------------------------------------------------------------
// Generic GEMM: out[tok][co] = sum_cin act[tok][cin] * W[cin][co] (+bias)
// The block's Kdim x 16 weight slab is DMA'd once by the Tensor Data Mover;
// K-loop reads W from LDS, act from global. Modes: 0=f32, 1=f16, 2=gelu->f16, 3=f32+res
// ---------------------------------------------------------------------------
__global__ void gemm_kernel(const h16* __restrict__ act, const h16* __restrict__ W,
                            const float* __restrict__ bias, float* __restrict__ outf,
                            h16* __restrict__ outh, const float* __restrict__ resid,
                            int Kdim, int N, int mode) {
  __shared__ h16 aw[512][16];  // up to Kdim=512 rows
  int wave = threadIdx.x >> 5, lane = threadIdx.x & 31, tid = threadIdx.x;
  int lg = (lane & 16) ? 1 : 0, ln16 = lane & 15;
  size_t t0 = (size_t)blockIdx.x * 128 + wave * 16;
  int Mb = blockIdx.y * 16;

  if (tid < 32) {  // wave 0 issues the TDM for the whole weight slab
    tdm_load_2d(W + Mb, (uint32_t)(uintptr_t)(&aw[0][0]),
                /*tensor_d0=*/16u + (uint32_t)Mb, /*tensor_d1=*/(uint32_t)Kdim,
                /*tile_d0=*/16u, /*tile_d1=*/(uint32_t)Kdim,
                /*stride0=*/(uint64_t)N);
    __builtin_amdgcn_s_wait_tensorcnt(0);
  }
  __syncthreads();

  v8f acc = {};
  for (int kb = 0; kb < Kdim; kb += 32) {
    if (kb + 32 < Kdim)
      __builtin_prefetch(act + (t0 + ln16) * (size_t)Kdim + kb + 32, 0, 0);
    v16h a, b;
#pragma unroll
    for (int j = 0; j < 16; ++j) a[j] = aw[kb + kmapA(lane, j)][ln16];
#pragma unroll
    for (int j = 0; j < 16; ++j)
      b[j] = act[(t0 + ln16) * (size_t)Kdim + kb + lg * 16 + j];
    acc = wmma16(a, b, acc);
  }
  if (bias) {
#pragma unroll
    for (int r = 0; r < 8; ++r) acc[r] += bias[Mb + r + lg * 8];
  }
  size_t tokn = t0 + ln16;
  int cob = Mb + lg * 8;
  size_t base = tokn * (size_t)N + cob;
  if (mode == 0 || mode == 3) {
    if (mode == 3) {
      const float4* rp = (const float4*)(resid + base);
      float4 r0 = rp[0], r1 = rp[1];
      acc[0] += r0.x; acc[1] += r0.y; acc[2] += r0.z; acc[3] += r0.w;
      acc[4] += r1.x; acc[5] += r1.y; acc[6] += r1.z; acc[7] += r1.w;
    }
    ((float4*)(outf + base))[0] = make_float4(acc[0], acc[1], acc[2], acc[3]);
    ((float4*)(outf + base))[1] = make_float4(acc[4], acc[5], acc[6], acc[7]);
  } else {
#pragma unroll
    for (int r = 0; r < 8; ++r) {
      float v = acc[r];
      if (mode == 2) v = 0.5f * v * (1.f + erff(v * 0.70710678118f));
      outh[base + r] = (h16)v;
    }
  }
}

// ---------------------------------------------------------------------------
// Flash windowed cross-attention. grid=(2, L=64, m=8), block=256 (8 waves).
// The (window, head) K/V slices (6 cams x 64 tok x 32 dh, f16) are DMA'd to
// LDS once per block by the TDM (3D descriptors), shared by all 8 q-tiles.
// ---------------------------------------------------------------------------
__global__ void attn_kernel(const h16* __restrict__ qp, const h16* __restrict__ kp,
                            const h16* __restrict__ vp, h16* __restrict__ attout) {
  __shared__ h16 ldsK[6 * 64 * 32];  // [cam*64+tok][dh] 24KB
  __shared__ h16 ldsV[6 * 64 * 32];  // 24KB
  __shared__ h16 ps[8][16][32];      // per-wave P round-trip
  int wave = threadIdx.x >> 5, lane = threadIdx.x & 31, tid = threadIdx.x;
  int qtile = blockIdx.x * 8 + wave;
  int l = blockIdx.y, m = blockIdx.z;
  int lg = (lane & 16) ? 1 : 0, ln16 = lane & 15;

  if (tid < 32) {  // wave 0: two 3D TDM loads (K and V), in-order per wave
    const h16* kbase = kp + (size_t)l * 64 * 256 + m * 32;
    const h16* vbase = vp + (size_t)l * 64 * 256 + m * 32;
    tdm_load_3d(kbase, (uint32_t)(uintptr_t)(&ldsK[0]), 32u, 64u, 6u,
                32u, 64u, 6u, /*stride0=*/256u, /*stride1=*/1048576u);
    tdm_load_3d(vbase, (uint32_t)(uintptr_t)(&ldsV[0]), 32u, 64u, 6u,
                32u, 64u, 6u, /*stride0=*/256u, /*stride1=*/1048576u);
    __builtin_amdgcn_s_wait_tensorcnt(0);
  }
  __syncthreads();

  v16h qa;
#pragma unroll
  for (int j = 0; j < 16; ++j)
    qa[j] = qp[(size_t)(l * 256 + qtile * 16 + ln16) * 256 + m * 32 + kmapA(lane, j)];

  v8f o0 = {}, o1 = {};
  float rmax[8], rsum[8];
#pragma unroll
  for (int r = 0; r < 8; ++r) {
    rmax[r] = -1e30f;
    rsum[r] = 0.f;
  }
  const float scale = 0.17677669529663687f;  // 1/sqrt(dh=32)

  for (int it = 0; it < 12; ++it) {
    int base = it * 32;  // local kv index within the LDS tile
    v16h kb1, kb2;
#pragma unroll
    for (int j = 0; j < 16; ++j) {
      int kk = lg * 16 + j;  // dh index (B K row)
      kb1[j] = ldsK[(base + ln16) * 32 + kk];
      kb2[j] = ldsK[(base + 16 + ln16) * 32 + kk];
    }
    v8f z = {};
    v8f s1 = wmma16(qa, kb1, z);
    v8f s2 = wmma16(qa, kb2, z);
#pragma unroll
    for (int r = 0; r < 8; ++r) {
      s1[r] *= scale;
      s2[r] *= scale;
      float cmax = redmax16(fmaxf(s1[r], s2[r]));
      float nmax = fmaxf(rmax[r], cmax);
      float corr = __expf(rmax[r] - nmax);
      rmax[r] = nmax;
      float p1 = __expf(s1[r] - nmax);
      float p2 = __expf(s2[r] - nmax);
      rsum[r] = rsum[r] * corr + redsum16(p1 + p2);
      o0[r] *= corr;
      o1[r] *= corr;
      int row = r + lg * 8;
      ps[wave][row][ln16] = (h16)p1;
      ps[wave][row][ln16 + 16] = (h16)p2;
    }
    v16h pa;
#pragma unroll
    for (int j = 0; j < 16; ++j) pa[j] = ps[wave][ln16][kmapA(lane, j)];
    v16h vb0, vb1;
#pragma unroll
    for (int j = 0; j < 16; ++j) {
      int kv = lg * 16 + j;  // kv index (B K row)
      vb0[j] = ldsV[(base + kv) * 32 + ln16];
      vb1[j] = ldsV[(base + kv) * 32 + 16 + ln16];
    }
    o0 = wmma16(pa, vb0, o0);
    o1 = wmma16(pa, vb1, o1);
  }
#pragma unroll
  for (int r = 0; r < 8; ++r) {
    float inv = 1.f / rsum[r];
    size_t row = (size_t)(l * 256 + qtile * 16 + r + lg * 8);
    attout[row * 256 + m * 32 + ln16] = (h16)(o0[r] * inv);
    attout[row * 256 + m * 32 + 16 + ln16] = (h16)(o1[r] * inv);
  }
}

// ---------------------------------------------------------------------------
// object_count sum, then weighted reduce over tokens -> out[256]
// ---------------------------------------------------------------------------
__global__ void ocsum_kernel(const float* __restrict__ oc, float* __restrict__ osum) {
  __shared__ float sp[8];
  int tid = threadIdx.x;
  float s = 0.f;
  for (int i = tid; i < 16384; i += 256) s += oc[i];
  s = redsum32(s);
  if ((tid & 31) == 0) sp[tid >> 5] = s;
  __syncthreads();
  if (tid == 0) {
    float t = 0.f;
    for (int i = 0; i < 8; ++i) t += sp[i];
    *osum = t;
  }
}

__global__ void reduce_kernel(const h16* __restrict__ lnp, const float* __restrict__ oc,
                              const float* __restrict__ osum, float* __restrict__ out) {
  __shared__ float sp[8];
  int d = blockIdx.x, tid = threadIdx.x;
  float s = 0.f;
  for (int tok = tid; tok < 16384; tok += 256) {
    int l = tok >> 8, qp2 = tok & 255;
    int h = (l >> 3) * 16 + (qp2 >> 4);
    int w = (l & 7) * 16 + (qp2 & 15);
    s += (float)lnp[(size_t)tok * 256 + d] * oc[h * 128 + w];
  }
  s = redsum32(s);
  if ((tid & 31) == 0) sp[tid >> 5] = s;
  __syncthreads();
  if (tid == 0) {
    float t = 0.f;
    for (int i = 0; i < 8; ++i) t += sp[i];
    out[d] = t / (*osum + 1e-5f);
  }
}

// ---------------------------------------------------------------------------
// Host-side orchestration
// ---------------------------------------------------------------------------
extern "C" void kernel_launch(void* const* d_in, const int* in_sizes, int n_in,
                              void* d_out, int out_size, void* d_ws, size_t ws_size,
                              hipStream_t stream) {
  (void)in_sizes; (void)n_in; (void)out_size; (void)ws_size;
  const int NQ = 16384;   // query tokens (L=64 windows * 256)
  const int NKT = 24576;  // kv tokens (6 cams * 64 windows * 64)

  const float* x = (const float*)d_in[0];
  const float* feature = (const float*)d_in[1];
  const float* oc = (const float*)d_in[4];
  // params flattened in dict insertion order starting at d_in[5]
  const float* P[53];
  for (int i = 0; i < 53; ++i) P[i] = (const float*)d_in[5 + i];
  // 0 bnp_g 1 bnp_b 2 bnp_m 3 bnp_v  4 bnl_g 5 bnl_b 6 bnl_m 7 bnl_v
  // 8 proj_w 9 lin_w 10 img_w
  // a1: 11 qln_g 12 qln_b 13 wq 14 bq 15 kln_g 16 kln_b 17 wk 18 bk
  //     19 vln_g 20 vln_b 21 wv 22 bv 23 wo 24 bo ; a2: 25..38 same
  // 39 pn1_g 40 pn1_b 41 w11 42 b11 43 w21 44 b21
  // 45 pn2_g 46 pn2_b 47 w12 48 b12 49 w22 50 b22  51 post_g 52 post_b

  char* wsb = (char*)d_ws;
  size_t off = 0;
  auto alloc = [&](size_t bytes) -> char* {
    char* p = wsb + off;
    off = (off + bytes + 255) & ~(size_t)255;
    return p;
  };
  float* keyT = (float*)alloc((size_t)NKT * 256 * 4);
  float* valT = (float*)alloc((size_t)NKT * 256 * 4);
  float* qA = (float*)alloc((size_t)NQ * 256 * 4);
  float* qB = (float*)alloc((size_t)NQ * 256 * 4);
  h16* lnQ = (h16*)alloc((size_t)NQ * 256 * 2);
  h16* lnK = (h16*)alloc((size_t)NKT * 256 * 2);
  h16* lnV = (h16*)alloc((size_t)NKT * 256 * 2);
  h16* qproj = (h16*)alloc((size_t)NQ * 256 * 2);
  h16* kproj = (h16*)alloc((size_t)NKT * 256 * 2);
  h16* vproj = (h16*)alloc((size_t)NKT * 256 * 2);
  h16* attout = (h16*)alloc((size_t)NQ * 256 * 2);
  h16* mlph = (h16*)alloc((size_t)NQ * 512 * 2);
  h16* projH = (h16*)alloc(65536 * 2);
  h16* imgH = (h16*)alloc(65536 * 2);
  h16* linH = (h16*)alloc(65536 * 2);
  h16* wq1H = (h16*)alloc(65536 * 2);
  h16* wk1H = (h16*)alloc(65536 * 2);
  h16* wv1H = (h16*)alloc(65536 * 2);
  h16* wo1H = (h16*)alloc(65536 * 2);
  h16* wq2H = (h16*)alloc(65536 * 2);
  h16* wk2H = (h16*)alloc(65536 * 2);
  h16* wv2H = (h16*)alloc(65536 * 2);
  h16* wo2H = (h16*)alloc(65536 * 2);
  h16* w11H = (h16*)alloc(131072 * 2);
  h16* w21H = (h16*)alloc(131072 * 2);
  h16* w12H = (h16*)alloc(131072 * 2);
  h16* w22H = (h16*)alloc(131072 * 2);
  float* invp = (float*)alloc(256 * 4);
  float* betap = (float*)alloc(256 * 4);
  float* invl = (float*)alloc(256 * 4);
  float* betal = (float*)alloc(256 * 4);
  float* ocsum = (float*)alloc(256);

  auto cvt = [&](const float* s, h16* d, int n) {
    cvt_kernel<<<(n + 255) / 256, 256, 0, stream>>>(s, d, n);
  };
  cvt(P[8], projH, 65536);
  cvt(P[10], imgH, 65536);
  cvt(P[9], linH, 65536);
  cvt(P[13], wq1H, 65536);
  cvt(P[17], wk1H, 65536);
  cvt(P[21], wv1H, 65536);
  cvt(P[23], wo1H, 65536);
  cvt(P[27], wq2H, 65536);
  cvt(P[31], wk2H, 65536);
  cvt(P[35], wv2H, 65536);
  cvt(P[37], wo2H, 65536);
  cvt(P[41], w11H, 131072);
  cvt(P[43], w21H, 131072);
  cvt(P[47], w12H, 131072);
  cvt(P[49], w22H, 131072);

  bnprep_kernel<<<1, 256, 0, stream>>>(P[0], P[1], P[2], P[3], P[4], P[5], P[6], P[7],
                                       invp, betap, invl, betal);

  feat_kernel<<<dim3(NKT / 16, 2), 256, 0, stream>>>(feature, projH, imgH, linH, invp,
                                                     betap, invl, betal, keyT, valT);
  gatherq_kernel<<<NQ, 256, 0, stream>>>(x, qA);

  auto gemm = [&](const h16* act, const h16* W, const float* bias, float* of, h16* oh,
                  const float* res, int T, int Kd, int N, int mode) {
    gemm_kernel<<<dim3(T / 128, N / 16), 256, 0, stream>>>(act, W, bias, of, oh, res,
                                                           Kd, N, mode);
  };

  // ---- attention layer 1 ----
  ln_kernel<<<NQ / 8, 256, 0, stream>>>(qA, P[11], P[12], lnQ, NQ);
  gemm(lnQ, wq1H, P[14], nullptr, qproj, nullptr, NQ, 256, 256, 1);
  ln_kernel<<<NKT / 8, 256, 0, stream>>>(keyT, P[15], P[16], lnK, NKT);
  gemm(lnK, wk1H, P[18], nullptr, kproj, nullptr, NKT, 256, 256, 1);
  ln_kernel<<<NKT / 8, 256, 0, stream>>>(valT, P[19], P[20], lnV, NKT);
  gemm(lnV, wv1H, P[22], nullptr, vproj, nullptr, NKT, 256, 256, 1);
  attn_kernel<<<dim3(2, 64, 8), 256, 0, stream>>>(qproj, kproj, vproj, attout);
  gemm(attout, wo1H, P[24], qA, nullptr, nullptr, NQ, 256, 256, 0);
  // MLP 1 (residual)
  ln_kernel<<<NQ / 8, 256, 0, stream>>>(qA, P[39], P[40], lnQ, NQ);
  gemm(lnQ, w11H, P[42], nullptr, mlph, nullptr, NQ, 256, 512, 2);
  gemm(mlph, w21H, P[44], qB, nullptr, qA, NQ, 512, 256, 3);

  // ---- attention layer 2 ----
  ln_kernel<<<NQ / 8, 256, 0, stream>>>(qB, P[25], P[26], lnQ, NQ);
  gemm(lnQ, wq2H, P[28], nullptr, qproj, nullptr, NQ, 256, 256, 1);
  ln_kernel<<<NKT / 8, 256, 0, stream>>>(keyT, P[29], P[30], lnK, NKT);
  gemm(lnK, wk2H, P[32], nullptr, kproj, nullptr, NKT, 256, 256, 1);
  ln_kernel<<<NKT / 8, 256, 0, stream>>>(valT, P[33], P[34], lnV, NKT);
  gemm(lnV, wv2H, P[36], nullptr, vproj, nullptr, NKT, 256, 256, 1);
  attn_kernel<<<dim3(2, 64, 8), 256, 0, stream>>>(qproj, kproj, vproj, attout);
  gemm(attout, wo2H, P[38], qA, nullptr, nullptr, NQ, 256, 256, 0);
  // MLP 2 (residual)
  ln_kernel<<<NQ / 8, 256, 0, stream>>>(qA, P[45], P[46], lnQ, NQ);
  gemm(lnQ, w12H, P[48], nullptr, mlph, nullptr, NQ, 256, 512, 2);
  gemm(mlph, w22H, P[50], qB, nullptr, qA, NQ, 512, 256, 3);

  // ---- post-LN + weighted reduce ----
  ln_kernel<<<NQ / 8, 256, 0, stream>>>(qB, P[51], P[52], lnQ, NQ);
  ocsum_kernel<<<1, 256, 0, stream>>>(oc, ocsum);
  reduce_kernel<<<256, 256, 0, stream>>>(lnQ, oc, ocsum, (float*)d_out);
}
